// QuantByteUnitLatent_59811714564598
// MI455X (gfx1250) — compile-verified, matrix operand order checked by reading, not run on hardware
//
#include <hip/hip_runtime.h>
#include <hip/hip_bf16.h>

typedef __attribute__((ext_vector_type(16))) _Float16 v16h;
typedef __attribute__((ext_vector_type(8)))  float    v8f;

#define N_ROWS 131072
#define HIDDEN 1024
#define LATENT 256

// workspace byte offsets
#define WS_A    0          // float[2]: a1, a2
#define WS_CQ1  64         // _Float16[8*1024] (W1 base codebook values, row-major [8][1024], exact in f16)
#define WS_MT   16448      // float[256*8]     (Mt = a1*a2*(c1@c2)^T)
#define WS_CQ2F 24640      // _Float16[16*32*32*16] (W2 base values, WMMA-B fragment order)

__device__ __forceinline__ float softplus_eps(float x) {
    return fmaxf(x, 0.0f) + log1pf(expf(-fabsf(x))) + 1e-6f;
}

__device__ __forceinline__ float quant_base(float w, float a, const float* cb) {
    // nearest level a*cb[i]; return the BASE value cb[i] (scale folded out later)
    float best = cb[0];
    float bd = fabsf(w - a * cb[0]);
#pragma unroll
    for (int i = 1; i < 7; ++i) {
        float d = fabsf(w - a * cb[i]);
        if (d < bd) { bd = d; best = cb[i]; }
    }
    return best;
}

__device__ __forceinline__ float fast_tanhf(float x) {
#if __has_builtin(__builtin_amdgcn_tanhf)
    return __builtin_amdgcn_tanhf(x);
#elif __has_builtin(__builtin_amdgcn_tanh_f32)
    return __builtin_amdgcn_tanh_f32(x);
#else
    float r;
    asm volatile("v_tanh_f32 %0, %1\n\tv_nop" : "=v"(r) : "v"(x));
    return r;
#endif
}

__device__ __forceinline__ float gelu_tanh(float x) {
    float x3 = x * x * x;
    float t = fast_tanhf(0.7978845608028654f * (x + 0.044715f * x3));
    return 0.5f * x * (1.0f + t);
}

// fragment slot math for cq2F: B[k][l] -> frag[ct][kc][lane][slot]
__device__ __forceinline__ int cq2f_index(int k, int l) {
    int ct = l >> 4, n = l & 15;
    int kc = k >> 5, kk = k & 31;
    int h    = (kk >> 3) & 1;
    int slot = (kk & 7) + 8 * (kk >> 4);
    int lane = n + 16 * h;
    return (((ct * 32 + kc) * 32) + lane) * 16 + slot;
}

// ---------------------------------------------------------------------------
// P0: scales only
// ---------------------------------------------------------------------------
__global__ void QBUL_scale_kernel(const float* __restrict__ a1raw,
                                  const float* __restrict__ a2raw,
                                  char* __restrict__ ws)
{
    if (threadIdx.x == 0) {
        float* wsA = (float*)(ws + WS_A);
        wsA[0] = softplus_eps(a1raw[0]);
        wsA[1] = softplus_eps(a2raw[0]);
    }
}

// ---------------------------------------------------------------------------
// P1: parallel quantization of W1 -> cq1 (f16) and W2 -> cq2F (f16, frag order)
// ---------------------------------------------------------------------------
__global__ __launch_bounds__(256) void QBUL_quant_kernel(
    const float* __restrict__ W1, const float* __restrict__ W2,
    const float* __restrict__ cbg, char* __restrict__ ws)
{
    const float* wsA  = (const float*)(ws + WS_A);
    _Float16*    cq1  = (_Float16*)(ws + WS_CQ1);
    _Float16*    cq2F = (_Float16*)(ws + WS_CQ2F);

    float cb[7];
#pragma unroll
    for (int i = 0; i < 7; ++i) cb[i] = cbg[i];
    const float a1 = wsA[0], a2 = wsA[1];

    const int gtid   = blockIdx.x * 256 + threadIdx.x;
    const int stride = gridDim.x * 256;

    for (int idx = gtid; idx < 8 * HIDDEN; idx += stride)
        cq1[idx] = (_Float16)quant_base(W1[idx], a1, cb);

    for (int idx = gtid; idx < HIDDEN * LATENT; idx += stride) {
        int k = idx / LATENT, l = idx % LATENT;
        cq2F[cq2f_index(k, l)] = (_Float16)quant_base(W2[idx], a2, cb);
    }
}

// ---------------------------------------------------------------------------
// P2: Mt[l][j] = a1*a2 * sum_k c1[j][k]*c2[k][l].  One block per l,
// deterministic tree reduction (no atomics -> bitwise reproducible).
// ---------------------------------------------------------------------------
__global__ __launch_bounds__(256) void QBUL_mt_kernel(char* __restrict__ ws)
{
    __shared__ float red[256];

    const float*    wsA  = (const float*)(ws + WS_A);
    const _Float16* cq1  = (const _Float16*)(ws + WS_CQ1);
    const _Float16* cq2F = (const _Float16*)(ws + WS_CQ2F);
    float*          Mt   = (float*)(ws + WS_MT);

    const int l   = blockIdx.x;
    const int tid = threadIdx.x;
    const float s12 = wsA[0] * wsA[1];

    float acc[8];
#pragma unroll
    for (int j = 0; j < 8; ++j) acc[j] = 0.0f;
#pragma unroll
    for (int ki = 0; ki < 4; ++ki) {
        int k = tid + 256 * ki;
        float c2 = (float)cq2F[cq2f_index(k, l)];
#pragma unroll
        for (int j = 0; j < 8; ++j)
            acc[j] = fmaf((float)cq1[j * HIDDEN + k], c2, acc[j]);
    }
#pragma unroll 1
    for (int j = 0; j < 8; ++j) {
        red[tid] = acc[j];
        __syncthreads();
        for (int s = 128; s > 0; s >>= 1) {
            if (tid < s) red[tid] += red[tid + s];
            __syncthreads();
        }
        if (tid == 0) Mt[l * 8 + j] = s12 * red[0];
        __syncthreads();
    }
}

// ---------------------------------------------------------------------------
// Main fused kernel: 16 rows per block, 256 threads = 8 wave32.
//   layer1 + gelu (VALU, hw tanh) -> hS f16 -> z GEMM (WMMA f16, fp32 acc,
//   ping-pong double-buffered K-loop) -> coalesced z out -> x_hat (VALU)
// ---------------------------------------------------------------------------
#define H_PITCH 1032   // 1024 + 8 halves: 16B row skew -> conflict-free ds_load_b128

__global__ __launch_bounds__(256) void QBUL_main_kernel(
    const float* __restrict__ x, const float* __restrict__ b1,
    const float* __restrict__ b2, const char* __restrict__ ws,
    float* __restrict__ z_out, float* __restrict__ xhat_out)
{
    __shared__ _Float16 hS[16 * H_PITCH];   // 33,024 B
    __shared__ float    xs[16 * 8];         //    512 B
    __shared__ float    bS[LATENT];         //  1,024 B
    __shared__ float    zS[16 * LATENT];    // 16,384 B

    const float*    wsA  = (const float*)(ws + WS_A);
    const _Float16* cq1  = (const _Float16*)(ws + WS_CQ1);
    const float*    Mt   = (const float*)(ws + WS_MT);
    const _Float16* cq2F = (const _Float16*)(ws + WS_CQ2F);

    const int tid  = threadIdx.x;
    const int row0 = blockIdx.x * 16;
    const float a1 = wsA[0], a2 = wsA[1];

    // stage x tile [16][8] and b2
    if (tid < 128) xs[tid] = x[(size_t)row0 * 8 + tid];
    bS[tid] = b2[tid];
    __syncthreads();

    // h = gelu(a1*(x @ c1) + b1), stored f16 row-major (pitch H_PITCH)
#pragma unroll 1
    for (int i = 0; i < 64; ++i) {
        int idx = tid + 256 * i;
        int m = idx >> 10, j = idx & 1023;
        float acc = 0.0f;
#pragma unroll
        for (int k = 0; k < 8; ++k)
            acc = fmaf(xs[m * 8 + k], (float)cq1[k * HIDDEN + j], acc);
        float hp = fmaf(a1, acc, b1[j]);
        hS[m * H_PITCH + j] = (_Float16)gelu_tanh(hp);
    }
    __syncthreads();

    // z = a2*(h @ c2) + b2 via v_wmma_f32_16x16x32_f16, K=1024 in 32 chunks
    const int wave = tid >> 5;
    const int lane = tid & 31;
    const int mrow = lane & 15;
    const int half = lane >> 4;
    const int ct0 = wave * 2, ct1 = wave * 2 + 1;

    union Frag { v16h h; int4 i4[2]; };

    const int aBase  = mrow * H_PITCH + 8 * half;      // LDS half-index
    const int bBase0 = (ct0 * 32 * 32 + lane) * 16;    // cq2F half-index
    const int bBase1 = (ct1 * 32 * 32 + lane) * 16;

#define LOAD_A(F, kc) do { const _Float16* p = &hS[aBase + (kc) * 32]; \
        (F).i4[0] = *(const int4*)(p); (F).i4[1] = *(const int4*)(p + 16); } while (0)
#define LOAD_B(F, base, kc) do { const _Float16* p = cq2F + (base) + (kc) * (32 * 16); \
        (F).i4[0] = *(const int4*)(p); (F).i4[1] = *(const int4*)(p + 8); } while (0)
#define WMMA(a, b, c) __builtin_amdgcn_wmma_f32_16x16x32_f16(false, (a).h, false, (b).h, \
                                                             (short)0, (c), false, false)

    v8f acc0 = {}, acc1 = {};
    Frag Aa, B0a, B1a, Ab, B0b, B1b;
    LOAD_A(Aa, 0); LOAD_B(B0a, bBase0, 0); LOAD_B(B1a, bBase1, 0);
    LOAD_A(Ab, 1); LOAD_B(B0b, bBase0, 1); LOAD_B(B1b, bBase1, 1);

#pragma unroll 1
    for (int kc = 0; kc < 30; kc += 2) {
        acc0 = WMMA(Aa, B0a, acc0);
        acc1 = WMMA(Aa, B1a, acc1);
        LOAD_A(Aa, kc + 2); LOAD_B(B0a, bBase0, kc + 2); LOAD_B(B1a, bBase1, kc + 2);
        acc0 = WMMA(Ab, B0b, acc0);
        acc1 = WMMA(Ab, B1b, acc1);
        LOAD_A(Ab, kc + 3); LOAD_B(B0b, bBase0, kc + 3); LOAD_B(B1b, bBase1, kc + 3);
    }
    acc0 = WMMA(Aa, B0a, acc0);
    acc1 = WMMA(Aa, B1a, acc1);
    acc0 = WMMA(Ab, B0b, acc0);
    acc1 = WMMA(Ab, B1b, acc1);

    // C/D layout: VGPR r -> M = r + 8*half, N = (lane&15) + 16*ct.  Stage to LDS.
    const int n0 = (lane & 15) + 16 * ct0;
    const int n1 = (lane & 15) + 16 * ct1;
#pragma unroll
    for (int r = 0; r < 8; ++r) {
        int m = r + 8 * half;
        zS[m * LATENT + n0] = fmaf(a2, acc0[r], bS[n0]);
        zS[m * LATENT + n1] = fmaf(a2, acc1[r], bS[n1]);
    }
    __syncthreads();

    // coalesced z_out stores: 16*256 floats = 1024 float4, 4 per thread
    {
        float4*       zo = (float4*)(z_out + (size_t)row0 * LATENT);
        const float4* zi = (const float4*)zS;
#pragma unroll
        for (int i = 0; i < 4; ++i) {
            int q = tid + 256 * i;
            zo[q] = zi[q];
        }
    }

    // x_hat = z @ Mt   (Mt already contains a1*a2*(c1@c2)^T)
    if (tid < 128) {
        int m = tid >> 3, j = tid & 7;
        float acc = 0.0f;
#pragma unroll 4
        for (int l = 0; l < LATENT; ++l)
            acc = fmaf(zS[m * LATENT + l], Mt[l * 8 + j], acc);
        xhat_out[(size_t)(row0 + m) * 8 + j] = acc;
    }
#undef LOAD_A
#undef LOAD_B
#undef WMMA
}

extern "C" void kernel_launch(void* const* d_in, const int* in_sizes, int n_in,
                              void* d_out, int out_size, void* d_ws, size_t ws_size,
                              hipStream_t stream)
{
    (void)in_sizes; (void)n_in; (void)out_size; (void)ws_size;
    const float* x   = (const float*)d_in[0];
    const float* W1  = (const float*)d_in[1];
    const float* W2  = (const float*)d_in[2];
    const float* a1r = (const float*)d_in[3];
    const float* a2r = (const float*)d_in[4];
    const float* b1  = (const float*)d_in[5];
    const float* b2  = (const float*)d_in[6];
    const float* cb  = (const float*)d_in[7];

    float* z_out    = (float*)d_out;                      // [N, 256]
    float* xhat_out = z_out + (size_t)N_ROWS * LATENT;    // [N, 8]
    char*  ws       = (char*)d_ws;

    QBUL_scale_kernel<<<1, 64, 0, stream>>>(a1r, a2r, ws);
    QBUL_quant_kernel<<<256, 256, 0, stream>>>(W1, W2, cb, ws);
    QBUL_mt_kernel<<<LATENT, 256, 0, stream>>>(ws);
    QBUL_main_kernel<<<N_ROWS / 16, 256, 0, stream>>>(x, b1, b2, ws, z_out, xhat_out);
}
